// SplitPart_8435315769729
// MI455X (gfx1250) — matrix operand order repaired and checked by reading, not compiled, not run on hardware
//
#include <hip/hip_runtime.h>

// Fused grouped tiny-MLP (3->16 lrelu, 16->4 + BN + lrelu) for 16 parts,
// mapped onto CDNA5 WMMA:
//   layer 1: v_wmma_f32_16x16x4_f32   (bias folded into K=3)
//   layer 2: v_wmma_f32_16x16x32_f16  (4 parts per tile, BN folded into A & C)
// One wave processes tiles of 16 consecutive f-positions (N dimension of WMMA).

#define EPSV 1e-5f

typedef __attribute__((ext_vector_type(2)))  float    v2f;
typedef __attribute__((ext_vector_type(8)))  float    v8f;
typedef __attribute__((ext_vector_type(2)))  _Float16 v2h;
typedef __attribute__((ext_vector_type(16))) _Float16 v16h;

__device__ __forceinline__ float lrelu(float x) { return fmaxf(x, 0.01f * x); }

// leaky-ReLU + f32->f16 pack done in packed-f16 ALU:
//   v_cvt_pk_f16_f32 ; v_pk_mul_f16 ; v_pk_max_num_f16
__device__ __forceinline__ v2h lrelu_pack(float a, float b) {
    v2h c;
    c[0] = (_Float16)a;
    c[1] = (_Float16)b;
    v2h scaled = c * (_Float16)0.01f;
    return __builtin_elementwise_max(c, scaled);
}

__global__ __launch_bounds__(256) void splitpart_wmma_kernel(
    const float* __restrict__ x,      // [256, 48, 2048]
    const float* __restrict__ W1,     // [16, 16, 3]
    const float* __restrict__ b1,     // [16, 16]
    const float* __restrict__ W2,     // [16, 4, 16]
    const float* __restrict__ b2,     // [16, 4]
    const float* __restrict__ gamma,  // [16, 4] (flat c = p*4+o)
    const float* __restrict__ beta,
    const float* __restrict__ rmean,
    const float* __restrict__ rvar,
    float* __restrict__ out)          // [256, 64, 2048]
{
    constexpr int F  = 2048;
    constexpr int Nn = 48;
    constexpr int C  = 64;
    constexpr int TOTAL_TILES = 256 * (F / 16);   // 32768
    constexpr int NWAVES      = 512 * 8;          // fixed launch: 512 blocks x 8 waves
    constexpr int ITERS       = TOTAL_TILES / NWAVES; // 8, exact

    const int lane = threadIdx.x & 31;
    const int mrow = lane & 15;       // M index (layer1: hidden; layer2: 4*p_local+o)
    const bool lo  = lane < 16;

    // ---------------- layer-1 A matrices: 16x4 f32, one per part -------------
    // A vgpr0: lanes<16 -> K=0 (W1[:, :, 0]) ; lanes>=16 -> K=2 (W1[:, :, 2])
    // A vgpr1: lanes<16 -> K=1 (W1[:, :, 1]) ; lanes>=16 -> K=3 (bias b1)
    v2f w1a[16];
#pragma unroll
    for (int p = 0; p < 16; ++p) {
        const float* wp = W1 + p * 48 + mrow * 3;
        w1a[p][0] = lo ? wp[0] : wp[2];
        w1a[p][1] = lo ? wp[1] : b1[p * 16 + mrow];
    }

    // ---------------- layer-2 A matrices: 16x32 f16, two per group of 4 parts
    // Group g covers parts q0..q3 = 4g..4g+3; output row m = 4*p_local + o.
    // K-map (matches lane-local B build from layer-1 D registers):
    //   K in [0,8)  -> (qlo, h=K)      K in [16,24) -> (qhi, h=K-16)
    //   K in [8,16) -> (qlo, h=K)    K in [24,32) -> (qhi, h=K-16)
    // A vgpr j: lanes<16 hold K = {2j, 2j+1} (j<4) / {16+2(j-4), +1} (j>=4)
    //           lanes>=16 hold K = {8+2j, +1} / {24+2(j-4), +1}
    //   -> j<4 part qlo, j>=4 part qhi in both lane halves; h = hh(j), which is +8 on lanes>=16.
    // BN scale s is folded into W2 here; BN shift folded into C init below.
    v16h  a2w[4][2];
    float accinit[4][8];
#pragma unroll
    for (int g = 0; g < 4; ++g) {
#pragma unroll
        for (int half = 0; half < 2; ++half) {    // half 0 -> rows 0-7, 1 -> rows 8-15
            v16h A;
#pragma unroll
            for (int j = 0; j < 8; ++j) {
                const int q  = 4 * g + 2 * half + (j >> 2);
                const int mb = 8 * half + 4 * (j >> 2);
                const bool valid = (mrow >= mb) && (mrow < mb + 4);
                const int o = (mrow - mb) & 3;
                const int c = q * 4 + o;
                const float s = gamma[c] * __frsqrt_rn(rvar[c] + EPSV);
                const int hh = 2 * (j & 3) + (lo ? 0 : 8);
                float w0 = W2[q * 64 + o * 16 + hh]     * s;
                float w1 = W2[q * 64 + o * 16 + hh + 1] * s;
                if (!valid) { w0 = 0.0f; w1 = 0.0f; }
                A[2 * j]     = (_Float16)w0;
                A[2 * j + 1] = (_Float16)w1;
            }
            a2w[g][half] = A;
        }
        // C accumulator init: s*(b2 - mean) + beta, per channel c = 16g + M
#pragma unroll
        for (int v = 0; v < 8; ++v) {
            const int c = 16 * g + v + (lo ? 0 : 8);
            const float s = gamma[c] * __frsqrt_rn(rvar[c] + EPSV);
            accinit[g][v] = s * (b2[c] - rmean[c]) + beta[c];
        }
    }

    // ---------------- uniform counted loop over 16-row tiles ------------------
    // Wave index forced scalar -> tile/base math in SALU, loop is s_cbranch.
    const int wv = __builtin_amdgcn_readfirstlane(
        (int)((blockIdx.x * blockDim.x + threadIdx.x) >> 5));
    const int noff = lo ? 0 : 2 * F;       // lanes>=16 load channel i=2 (K=2)

#pragma unroll 1
    for (int i = 0; i < ITERS; ++i) {
        const int t    = wv + i * NWAVES;
        const int bidx = t >> 7;               // F/16 = 128 tiles per batch image
        const int f0   = (t & 127) << 4;
        const float* xb = x + (size_t)bidx * Nn * F + f0 + mrow;
        float*       ob = out + (size_t)bidx * C * F + f0 + mrow;

#pragma unroll
        for (int g = 0; g < 4; ++g) {
            // ---- layer 1: four parts, one f32 WMMA each ----
            v8f H[4];
#pragma unroll
            for (int pl = 0; pl < 4; ++pl) {
                const int p = 4 * g + pl;
                // B vgpr0: K=0 (lo) / K=2 (hi); B vgpr1: K=1 (lo) / K=3 == 1.0 (hi)
                v2f bb;
                bb[0] = xb[(3 * p) * F + noff];
                bb[1] = lo ? xb[(3 * p + 1) * F] : 1.0f;
                v8f z = {};
                H[pl] = __builtin_amdgcn_wmma_f32_16x16x4_f32(
                            false, w1a[p], false, bb, (short)0, z, false, false);
            }
            // ---- lrelu + lane-local f32->f16 pack into layer-2 B matrices ----
            // B vgpr j (all lanes) = pack(H[src][2(j&3)], H[src][2(j&3)+1]):
            //   lanes<16 supply K in [0,8)+[16,24) (part q_src), lanes>=16 K in [8,16)+[24,32).
            v16h B1, B2;
#pragma unroll
            for (int j = 0; j < 8; ++j) {
                const int src = j >> 2;            // H[0]/H[1] for B1, H[2]/H[3] for B2
                const int e   = 2 * (j & 3);
                v2h r1 = lrelu_pack(H[src][e],     H[src][e + 1]);
                v2h r2 = lrelu_pack(H[2 + src][e], H[2 + src][e + 1]);
                B1[2 * j]     = r1[0];
                B1[2 * j + 1] = r1[1];
                B2[2 * j]     = r2[0];
                B2[2 * j + 1] = r2[1];
            }
            // ---- layer 2: two chained f16 WMMAs, BN pre-folded ----
            v8f acc;
#pragma unroll
            for (int v = 0; v < 8; ++v) acc[v] = accinit[g][v];
            acc = __builtin_amdgcn_wmma_f32_16x16x32_f16(
                      false, a2w[g][0], false, B1, (short)0, acc, false, false);
            acc = __builtin_amdgcn_wmma_f32_16x16x32_f16(
                      false, a2w[g][1], false, B2, (short)0, acc, false, false);
            // ---- final lrelu + coalesced non-temporal stores ----
#pragma unroll
            for (int v = 0; v < 8; ++v) {
                const int c = 16 * g + v + (lo ? 0 : 8);
                __builtin_nontemporal_store(lrelu(acc[v]), &ob[(size_t)c * F]);
            }
        }
    }
}

extern "C" void kernel_launch(void* const* d_in, const int* in_sizes, int n_in,
                              void* d_out, int out_size, void* d_ws, size_t ws_size,
                              hipStream_t stream) {
    (void)in_sizes; (void)n_in; (void)d_ws; (void)ws_size; (void)out_size;
    const float* x     = (const float*)d_in[0];
    const float* W1    = (const float*)d_in[1];
    const float* b1    = (const float*)d_in[2];
    const float* W2    = (const float*)d_in[3];
    const float* b2    = (const float*)d_in[4];
    const float* gamma = (const float*)d_in[5];
    const float* beta  = (const float*)d_in[6];
    const float* rmean = (const float*)d_in[7];
    const float* rvar  = (const float*)d_in[8];
    float* out = (float*)d_out;

    // 512 blocks x 8 waves = 4096 waves; 32768 tiles -> 8 tiles per wave (exact).
    splitpart_wmma_kernel<<<512, 256, 0, stream>>>(
        x, W1, b1, W2, b2, gamma, beta, rmean, rvar, out);
}
